// EEGSNN2_8400956031529
// MI455X (gfx1250) — matrix operand order repaired and checked by reading, not compiled, not run on hardware
//
#include <hip/hip_runtime.h>

// ---------- types ----------
typedef __bf16 bf16;
typedef __attribute__((ext_vector_type(16))) __bf16 bf16x16;
typedef __attribute__((ext_vector_type(8)))  __bf16 bf16x8;
typedef __attribute__((ext_vector_type(8)))  float  f32x8;
typedef __attribute__((ext_vector_type(4)))  float  f32x4;

union BFrag { bf16x16 v16; bf16x8 v8[2]; };

// ---------- problem sizes ----------
#define BATCH   256
#define TSTEPS  64
#define K1      4096         // 32*128
#define N1      512
#define K2      512
#define N2      40
#define N2P     48           // padded to 3 WMMA tiles
#define M_TOT   (BATCH*TSTEPS)   // 16384

__device__ __forceinline__ f32x8 zero8() {
    f32x8 z = {0.f,0.f,0.f,0.f,0.f,0.f,0.f,0.f};
    return z;
}

// LDS byte offset of a __shared__ pointer: generic-address low 32 bits are the
// LDS offset on CDNA5 (aperture mapping truncates to addr[31:0]).
__device__ __forceinline__ unsigned lds_off(const void* p) {
    return (unsigned)(unsigned long long)(uintptr_t)p;
}

// ---------- weight transpose/convert ----------
// W1 [4096][512] f32 -> W1T [512][4096] bf16 (write-coalesced)
__global__ void eeg_w1t(const float* __restrict__ W1, bf16* __restrict__ W1T) {
    int tid = blockIdx.x * blockDim.x + threadIdx.x;  // 512*4096 threads
    int n = tid >> 12, k = tid & 4095;
    W1T[tid] = (bf16)W1[k * N1 + n];
}

// W2 [512][40] f32 -> W2T [48][512] bf16, cols 40..47 zero
__global__ void eeg_w2t(const float* __restrict__ W2, bf16* __restrict__ W2T) {
    int tid = blockIdx.x * blockDim.x + threadIdx.x;  // 48*512 threads
    int n = tid >> 9, k = tid & 511;
    float v = (n < N2) ? W2[k * N2 + n] : 0.f;
    W2T[tid] = (bf16)v;
}

// ---------- GEMM1: Y1[16384,512] = bf16(X)[16384,4096] @ bf16(W1) + b1 ----------
// Workgroup tile 128M x 256N, 8 waves (2M x 4N), wave tile 64x64 = 16 WMMA accums.
// A (f32 -> bf16 convert) staged through 16 VGPRs; B (already bf16) staged with
// CDNA5 async-to-LDS (no VGPR data, ASYNCcnt-tracked, overlaps WMMA).
#define APITCH 40
#define BPITCH 40

__global__ __launch_bounds__(256)
void eeg_gemm1(const float* __restrict__ X, const bf16* __restrict__ W1T,
               const float* __restrict__ b1, float* __restrict__ Y1) {
    __shared__ bf16 As[2][128 * APITCH];   // 2 * 10240 B
    __shared__ bf16 Bs[2][256 * BPITCH];   // 2 * 20480 B

    const int tid  = threadIdx.x;
    const int mbase = blockIdx.x * 128;
    const int nbase = blockIdx.y * 256;
    const int w = tid >> 5, lane = tid & 31;
    const int wm = w >> 2, wn = w & 3;       // 2 x 4 wave grid
    const int hi = lane >> 4, l15 = lane & 15;

    // staging roles: A -> thread covers (row, 16 cols); B -> thread covers one W1T row slab
    const int ar = tid >> 1;
    const int ac = (tid & 1) * 16;
    const float* aSrc = X + (size_t)(mbase + ar) * K1 + ac;
    const bf16*  bSrc = W1T + (size_t)(nbase + tid) * K1;
    const unsigned bLds0 = lds_off(&Bs[0][tid * BPITCH]);
    const unsigned bLds1 = lds_off(&Bs[1][tid * BPITCH]);

    f32x8 acc[4][4];
#pragma unroll
    for (int i = 0; i < 4; ++i)
#pragma unroll
        for (int j = 0; j < 4; ++j) acc[i][j] = zero8();

    f32x4 fa[4];

    auto load_A_regs = [&](int kb) {
        const float* pa = aSrc + kb;
#pragma unroll
        for (int i = 0; i < 4; ++i)
            fa[i] = *reinterpret_cast<const f32x4*>(pa + 4 * i);
        if (kb + 64 <= K1)                        // hint the next K-slab of X
            __builtin_prefetch(pa + 32, 0, 1);    // -> global_prefetch_b8
    };
    auto store_A_lds = [&](int buf) {
        bf16x8 h0, h1;
#pragma unroll
        for (int i = 0; i < 4; ++i) {
            h0[i]     = (bf16)fa[0][i];
            h0[4 + i] = (bf16)fa[1][i];
            h1[i]     = (bf16)fa[2][i];
            h1[4 + i] = (bf16)fa[3][i];
        }
        *reinterpret_cast<bf16x8*>(&As[buf][ar * APITCH + ac])     = h0;
        *reinterpret_cast<bf16x8*>(&As[buf][ar * APITCH + ac + 8]) = h1;
    };
    // async byte-copy of one W1T row slab (64B) into LDS: 4 x B128 per thread
    auto async_B = [&](int buf, int kb) {
        const unsigned lbase = buf ? bLds1 : bLds0;
#pragma unroll
        for (int i = 0; i < 4; ++i) {
            unsigned long long ga =
                (unsigned long long)(uintptr_t)(bSrc + kb + 8 * i);
            unsigned ldst = lbase + 16u * i;
            asm volatile("global_load_async_to_lds_b128 %0, %1, off"
                         :: "v"(ldst), "v"(ga) : "memory");
        }
    };

    async_B(0, 0);
    load_A_regs(0);
    int buf = 0;
    const int NS = K1 / 32;                  // 128 K-steps
    for (int ks = 0; ks < NS; ++ks) {
        store_A_lds(buf);
        asm volatile("s_wait_asynccnt 0x0" ::: "memory");  // B(buf) landed
        __syncthreads();                                   // all staging visible
        if (ks + 1 < NS) {
            async_B(buf ^ 1, (ks + 1) * 32);   // fire-and-forget over WMMA
            load_A_regs((ks + 1) * 32);        // A loads in flight over WMMA
        }

        BFrag bfr[4];
#pragma unroll
        for (int j = 0; j < 4; ++j) {
            const int bb = (wn * 64 + j * 16 + l15) * BPITCH + hi * 8;
            bfr[j].v8[0] = *reinterpret_cast<const bf16x8*>(&Bs[buf][bb]);
            bfr[j].v8[1] = *reinterpret_cast<const bf16x8*>(&Bs[buf][bb + 16]);
        }
#pragma unroll
        for (int i = 0; i < 4; ++i) {
            BFrag afr;
            const int ab = (wm * 64 + i * 16 + l15) * APITCH + hi * 8;
            afr.v8[0] = *reinterpret_cast<const bf16x8*>(&As[buf][ab]);
            afr.v8[1] = *reinterpret_cast<const bf16x8*>(&As[buf][ab + 16]);
#pragma unroll
            for (int j = 0; j < 4; ++j)
                acc[i][j] = __builtin_amdgcn_wmma_f32_16x16x32_bf16(
                    false, afr.v16, false, bfr[j].v16, (short)0, acc[i][j],
                    false, false);
        }
        buf ^= 1;
    }

    // epilogue: C layout = VGPR v -> row (v + 8*hi), lane&15 -> col
#pragma unroll
    for (int j = 0; j < 4; ++j) {
        const int col = nbase + wn * 64 + j * 16 + l15;
        const float bias = b1[col];
#pragma unroll
        for (int i = 0; i < 4; ++i) {
            const int row0 = mbase + wm * 64 + i * 16 + hi * 8;
#pragma unroll
            for (int v = 0; v < 8; ++v)
                Y1[(size_t)(row0 + v) * N1 + col] = acc[i][j][v] + bias;
        }
    }
}

// ---------- LIF layer 1: sequential over T per (b,n), spikes -> bf16 ----------
__global__ void eeg_lif1(const float* __restrict__ Y1, bf16* __restrict__ S1) {
    const int tid = blockIdx.x * blockDim.x + threadIdx.x;  // 256*512 threads
    const int b = tid >> 9, n = tid & 511;
    float mem = 0.f, spk = 0.f;
    const int rbase = b * TSTEPS;
#pragma unroll 4
    for (int t = 0; t < TSTEPS; ++t) {
        const int idx = (rbase + t) * N1 + n;
        mem = mem * 0.5f * (1.f - spk) + Y1[idx];
        spk = (mem - 1.f > 0.f) ? 1.f : 0.f;
        S1[idx] = (bf16)spk;
    }
}

// ---------- GEMM2: Y2[16384,48] = S1[16384,512] @ W2T^T + b2 (register-direct) ----------
__global__ __launch_bounds__(256)
void eeg_gemm2(const bf16* __restrict__ S1, const bf16* __restrict__ W2T,
               const float* __restrict__ b2, float* __restrict__ Y2) {
    const int tid = threadIdx.x, w = tid >> 5, lane = tid & 31;
    const int hi = lane >> 4, l15 = lane & 15;
    const int mbase = (blockIdx.x * 8 + w) * 16;

    f32x8 acc[3];
#pragma unroll
    for (int j = 0; j < 3; ++j) acc[j] = zero8();

    const bf16* aRow = S1 + (size_t)(mbase + l15) * K2;
#pragma unroll
    for (int ks = 0; ks < K2 / 32; ++ks) {
        const int kb = ks * 32;
        BFrag afr;
        afr.v8[0] = *reinterpret_cast<const bf16x8*>(aRow + kb + hi * 8);
        afr.v8[1] = *reinterpret_cast<const bf16x8*>(aRow + kb + 16 + hi * 8);
#pragma unroll
        for (int j = 0; j < 3; ++j) {
            const bf16* bRow = W2T + (size_t)(j * 16 + l15) * K2 + kb + hi * 8;
            BFrag bfr;
            bfr.v8[0] = *reinterpret_cast<const bf16x8*>(bRow);
            bfr.v8[1] = *reinterpret_cast<const bf16x8*>(bRow + 16);
            acc[j] = __builtin_amdgcn_wmma_f32_16x16x32_bf16(
                false, afr.v16, false, bfr.v16, (short)0, acc[j], false, false);
        }
    }
#pragma unroll
    for (int j = 0; j < 3; ++j) {
        const int col = j * 16 + l15;
        const float bias = (col < N2) ? b2[col] : 0.f;
        const int row0 = mbase + hi * 8;
#pragma unroll
        for (int v = 0; v < 8; ++v)
            Y2[(size_t)(row0 + v) * N2P + col] = acc[j][v] + bias;
    }
}

// ---------- LIF layer 2 + AvgPool(10) + time mean -> out[256,4] ----------
__global__ void eeg_lif2pool(const float* __restrict__ Y2, float* __restrict__ out) {
    const int tid = blockIdx.x * blockDim.x + threadIdx.x;  // 1024 threads
    const int b = tid >> 2, g = tid & 3;
    float mem[10], spk[10];
#pragma unroll
    for (int q = 0; q < 10; ++q) { mem[q] = 0.f; spk[q] = 0.f; }
    float s = 0.f;
    for (int t = 0; t < TSTEPS; ++t) {
        const float* p = Y2 + (size_t)(b * TSTEPS + t) * N2P + g * 10;
#pragma unroll
        for (int q = 0; q < 10; ++q) {
            mem[q] = mem[q] * 0.5f * (1.f - spk[q]) + p[q];
            spk[q] = (mem[q] - 1.f > 0.f) ? 1.f : 0.f;
            s += spk[q];
        }
    }
    out[tid] = s * (1.f / (float)(TSTEPS * 10));
}

// ---------- launch ----------
extern "C" void kernel_launch(void* const* d_in, const int* in_sizes, int n_in,
                              void* d_out, int out_size, void* d_ws, size_t ws_size,
                              hipStream_t stream) {
    const float* X  = (const float*)d_in[0];
    const float* W1 = (const float*)d_in[1];
    const float* b1 = (const float*)d_in[2];
    const float* W2 = (const float*)d_in[3];
    const float* b2 = (const float*)d_in[4];
    float* out = (float*)d_out;

    char* ws = (char*)d_ws;
    // workspace layout (bytes), all 16B aligned, total ~54.9 MB
    bf16*  W1T = (bf16*)(ws);                                   //  4,194,304
    bf16*  W2T = (bf16*)(ws + 4194304);                         //     49,152
    float* Y1  = (float*)(ws + 4194304 + 49152);                // 33,554,432
    bf16*  S1  = (bf16*)(ws + 4194304 + 49152 + 33554432);      // 16,777,216
    float* Y2  = (float*)(ws + 4194304 + 49152 + 33554432 + 16777216); // 3,145,728

    eeg_w1t<<<(N1 * K1) / 256, 256, 0, stream>>>(W1, W1T);
    eeg_w2t<<<(N2P * K2) / 256, 256, 0, stream>>>(W2, W2T);
    eeg_gemm1<<<dim3(M_TOT / 128, N1 / 256), 256, 0, stream>>>(X, W1T, b1, Y1);
    eeg_lif1<<<(BATCH * N1) / 256, 256, 0, stream>>>(Y1, S1);
    eeg_gemm2<<<M_TOT / 128, 256, 0, stream>>>(S1, W2T, b2, Y2);
    eeg_lif2pool<<<(BATCH * 4) / 256, 256, 0, stream>>>(Y2, out);
}
